// FLAGatedLinearAttention_89498528514090
// MI455X (gfx1250) — compile-verified, matrix operand order checked by reading, not run on hardware
//
#include <hip/hip_runtime.h>
#include <hip/hip_bf16.h>

// ---------------------------------------------------------------------------
// GLA forward for MI455X (gfx1250, wave32, WMMA + TDM).
// Shapes: B=4 T=4096 D=768 H=12 K=384 V=768 HK=32 HV=64 LR=16
// ---------------------------------------------------------------------------

typedef __attribute__((ext_vector_type(16))) __bf16 v16bf;
typedef __attribute__((ext_vector_type(8)))  __bf16 v8bf;
typedef __attribute__((ext_vector_type(4)))  __bf16 v4bf;
typedef __attribute__((ext_vector_type(2)))  __bf16 v2bf;
typedef __attribute__((ext_vector_type(8)))  float  v8f;
typedef __attribute__((ext_vector_type(4)))  float  v4f;
typedef __attribute__((ext_vector_type(2)))  float  v2f;
typedef __attribute__((ext_vector_type(4)))  unsigned int u32x4;
typedef __attribute__((ext_vector_type(4)))  int    i32x4;
typedef __attribute__((ext_vector_type(8)))  int    i32x8;

#define B_   4
#define T_   4096
#define D_   768
#define H_   12
#define K_   384
#define V_   768
#define HK_  32
#define HV_  64
#define NTOK (B_ * T_)
#define SCALE_ 0.17677669529663687f   // 32^-0.5
#define EPS_   1e-5f
#define GLN_   16.0f

// TDM availability (device pass only)
#if defined(__HIP_DEVICE_COMPILE__) && defined(__gfx1250__) && defined(__has_builtin)
#if __has_builtin(__builtin_amdgcn_tensor_load_to_lds) && __has_builtin(__builtin_amdgcn_s_wait_tensorcnt)
#define HAVE_TDM 1
#endif
#endif
#ifndef HAVE_TDM
#define HAVE_TDM 0
#endif

__device__ __forceinline__ v16bf concat8(v8bf lo, v8bf hi) {
  return __builtin_shufflevector(lo, hi, 0, 1, 2, 3, 4, 5, 6, 7,
                                         8, 9, 10, 11, 12, 13, 14, 15);
}

#if HAVE_TDM
// Issue one 2D TDM tile load: group0 carries lds/global addresses, group1 the
// (pre-built, loop-invariant) tile/tensor geometry.
__device__ __forceinline__ void tdm_load_2d(unsigned ldsOff,
                                            unsigned long long ga, i32x8 g1) {
  u32x4 g0;
  g0[0] = 1u;                                            // count=1 descriptor
  g0[1] = ldsOff;                                        // lds_addr
  g0[2] = (unsigned)ga;                                  // global_addr lo
  g0[3] = ((unsigned)(ga >> 32) & 0x01FFFFFFu) | (2u << 30); // hi | type=2
  i32x4 z4 = {};
#if defined(__clang_major__) && (__clang_major__ >= 23)
  i32x8 z8 = {};
  __builtin_amdgcn_tensor_load_to_lds(g0, g1, z4, z4, z8, 0);
#else
  __builtin_amdgcn_tensor_load_to_lds(g0, g1, z4, z4, 0);
#endif
}
#endif

// ---------------------------------------------------------------------------
// Tiled GEMM: C[M,N] = alpha * A[M,Kd] @ W[Kd,N]   (fp32 in, bf16 WMMA, fp32 acc)
// Block = 256 threads = 8 waves. Tile BM=64, BN=128, BK=32.
// A staged row-major, W staged TRANSPOSED so every WMMA fragment is two
// aligned 16-byte ds_load_b128 + one register concat. Staging issues all
// global loads as one clause, then converts: transpose stores are packed
// bf16 row-pairs (single v_cvt_pk_bf16_f32 + ds_store_b32 each).
// ---------------------------------------------------------------------------
__global__ __launch_bounds__(256) void gemm_bf16_wmma(
    const float* __restrict__ A, const float* __restrict__ W,
    float* __restrict__ C, int M, int N, int Kd, float alpha)
{
  __shared__ __bf16 As[64][40];     // row stride 80B: 16B-aligned, conflict-free
  __shared__ __bf16 Bst[128][40];   // transposed: Bst[n][k]

  const int bm = blockIdx.y * 64;
  const int bn = blockIdx.x * 128;
  const int tid  = threadIdx.x;
  const int wave = tid >> 5;
  const int lane = tid & 31;
  const int msub  = wave >> 1;
  const int nbase = (wave & 1) * 64;

  v8f acc[4];
#pragma unroll
  for (int i = 0; i < 4; ++i) acc[i] = (v8f){};

  const int ar = tid >> 2;          // A stage: row 0..63
  const int ac = (tid & 3) * 8;     // A stage: col 0,8,16,24

  for (int kt = 0; kt < Kd; kt += 32) {
    // ---- issue ALL staging loads for this K-step up front (one clause) ----
    const float* asrc = &A[(size_t)(bm + ar) * Kd + kt + ac];
    v4f a0 = *(const v4f*)asrc;
    v4f a1 = *(const v4f*)(asrc + 4);
    v4f w0[2], w1[2];
#pragma unroll
    for (int e = 0; e < 2; ++e) {
      const int idx = tid + 256 * e;     // 0..511 micro-tiles (2 rows x 4 cols)
      const int rp  = idx >> 5;          // row pair 0..15 -> k rows 2*rp, 2*rp+1
      const int c   = (idx & 31) * 4;    // n col 0..124
      const float* wsrc = &W[(size_t)(kt + rp * 2) * N + bn + c];
      w0[e] = *(const v4f*)wsrc;
      w1[e] = *(const v4f*)(wsrc + N);
    }
    if (kt + 32 < Kd)                    // global_prefetch_b8 for next A tile
      __builtin_prefetch(&A[(size_t)(bm + ar) * Kd + kt + 32 + ac], 0, 1);

    // ---- convert + stage A (two packed b64 stores) ----
    {
      v4bf h0, h1;
#pragma unroll
      for (int j = 0; j < 4; ++j) { h0[j] = (__bf16)a0[j]; h1[j] = (__bf16)a1[j]; }
      *(v4bf*)&As[ar][ac]     = h0;
      *(v4bf*)&As[ar][ac + 4] = h1;
    }
    // ---- convert + stage W transposed (packed row-pair b32 stores) ----
#pragma unroll
    for (int e = 0; e < 2; ++e) {
      const int idx = tid + 256 * e;
      const int rp  = idx >> 5;
      const int c   = (idx & 31) * 4;
#pragma unroll
      for (int j = 0; j < 4; ++j) {
        v2bf p;
        p[0] = (__bf16)w0[e][j];
        p[1] = (__bf16)w1[e][j];
        *(v2bf*)&Bst[c + j][rp * 2] = p;   // 4B-aligned packed store
      }
    }
    __syncthreads();

    // fragments: lane half selects K bands {0..7,16..23} / {8..15,24..31}
    const int m   = msub * 16 + (lane & 15);
    const int kb0 = (lane < 16) ? 0 : 8;
    v8bf alo = *(const v8bf*)&As[m][kb0];
    v8bf ahi = *(const v8bf*)&As[m][kb0 + 16];
    v16bf afrag = concat8(alo, ahi);
#pragma unroll
    for (int ns = 0; ns < 4; ++ns) {
      const int n = nbase + ns * 16 + (lane & 15);
      v8bf blo = *(const v8bf*)&Bst[n][kb0];
      v8bf bhi = *(const v8bf*)&Bst[n][kb0 + 16];
      v16bf bfrag = concat8(blo, bhi);
      acc[ns] = __builtin_amdgcn_wmma_f32_16x16x32_bf16(
          false, afrag, false, bfrag, (short)0, acc[ns], false, false);
    }
    __syncthreads();
  }

  // epilogue: C/D layout -> lane = col, rows = vgpr + 8*(lane>=16)
  const int rb = (lane < 16) ? 0 : 8;
#pragma unroll
  for (int ns = 0; ns < 4; ++ns) {
    const int col = bn + nbase + ns * 16 + (lane & 15);
#pragma unroll
    for (int r2 = 0; r2 < 8; ++r2) {
      const int row = bm + msub * 16 + rb + r2;
      if (row < M && col < N) C[(size_t)row * N + col] = alpha * acc[ns][r2];
    }
  }
}

// ---------------------------------------------------------------------------
// gk = logsigmoid((x @ Wgk1) @ Wgk2 + b) / 16 ; one block per token.
// Rank-16 inner product reduced with 16-lane shuffle trees (wave32 native).
// ---------------------------------------------------------------------------
__global__ __launch_bounds__(256) void gk_lowrank_kernel(
    const float* __restrict__ x, const float* __restrict__ Wgk1,
    const float* __restrict__ Wgk2, const float* __restrict__ bgk2,
    float* __restrict__ gk)
{
  __shared__ float xs[D_];
  __shared__ float h[16];
  const int row = blockIdx.x;
  const int tid = threadIdx.x;

  for (int i = tid; i < D_; i += 256) xs[i] = x[(size_t)row * D_ + i];
  __syncthreads();

  // 16 groups of 16 lanes; group g owns hidden unit g
  const int g = tid >> 4;
  const int t = tid & 15;
  float acc = 0.f;
  for (int i = t; i < D_; i += 16) acc += xs[i] * Wgk1[i * 16 + g];
#pragma unroll
  for (int off = 8; off; off >>= 1) acc += __shfl_xor(acc, off, 16);
  if (t == 0) h[g] = acc;
  __syncthreads();

  for (int j = tid; j < K_; j += 256) {
    float a2 = bgk2[j];
#pragma unroll
    for (int r = 0; r < 16; ++r) a2 += h[r] * Wgk2[r * K_ + j];
    const float ls = fminf(a2, 0.f) - log1pf(__expf(-fabsf(a2)));
    gk[(size_t)row * K_ + j] = ls * (1.0f / GLN_);
  }
}

// ---------------------------------------------------------------------------
// Chunked GLA scan. One block per (b,h); 8 waves; chunk C=16.
// q/k/g/v chunk tiles DMA'd by the Tensor Data Mover (4 tensor_load_to_lds +
// one s_wait_tensorcnt per chunk); q tile uses TDM LDS padding to land in the
// bank-conflict-free qe[16][34] layout directly.
// State S[32][64] lives in per-wave v8f accumulators (wave w owns tile
// (w>>2, w&3)); mirrored to LDS once per chunk for the q@S_prev readout.
// ---------------------------------------------------------------------------
__global__ __launch_bounds__(256) void gla_scan_kernel(
    const float* __restrict__ qb, const float* __restrict__ kb,
    const float* __restrict__ vb, const float* __restrict__ gkb,
    float* __restrict__ ob, int Tlen)
{
  const int bh = blockIdx.x;
  const int b  = bh / H_;
  const int h  = bh % H_;

  __shared__ float  qe[16][34];      // q * exp(Gcum) * scale (fp32), padded
  __shared__ __bf16 qe_bf[16][32];
  __shared__ __bf16 ke_bf[16][32];   // k * exp(-Gcum)
  __shared__ float  keu[16][32];     // k * exp(Gtot - Gcum)
  __shared__ float  gcs[16][32];     // cumulative gates
  __shared__ float  vch[16][64];     // contiguous -> TDM tile target
  __shared__ float  a_s[16][18];     // masked intra-chunk scores, padded
  __shared__ float  S_s[32][64];     // S_prev mirror
  __shared__ float  eTot[32];        // exp(Gtot)

  const int tid  = threadIdx.x;
  const int wave = tid >> 5;
  const int lane = tid & 31;
  const int mt = wave >> 2, nt = wave & 3;

  for (int i = tid; i < 32 * 64; i += 256) (&S_s[0][0])[i] = 0.f;
  v8f sacc = {};
  __syncthreads();

#if HAVE_TDM
  // loop-invariant descriptor geometry (group 1)
  i32x8 g1q;  // K-side tiles: dim0=384, tile 32x16, pad 2 dwords / 32 dwords
  g1q[0] = (int)((2u << 16) | (1u << 20) | (4u << 22) | (1u << 25));
  g1q[1] = (int)(((unsigned)K_ & 0xFFFFu) << 16);
  g1q[2] = (int)(((unsigned)NTOK & 0xFFFFu) << 16);
  g1q[3] = (int)((((unsigned)NTOK >> 16) & 0xFFFFu) | (32u << 16));
  g1q[4] = 16; g1q[5] = K_; g1q[6] = 0; g1q[7] = 0;
  i32x8 g1k = g1q;                 // same geometry, no LDS padding
  g1k[0] = (int)(2u << 16);
  i32x8 g1v;                       // V-side tile: dim0=768, tile 64x16
  g1v[0] = (int)(2u << 16);
  g1v[1] = (int)(((unsigned)V_ & 0xFFFFu) << 16);
  g1v[2] = (int)(((unsigned)NTOK & 0xFFFFu) << 16);
  g1v[3] = (int)((((unsigned)NTOK >> 16) & 0xFFFFu) | (64u << 16));
  g1v[4] = 16; g1v[5] = V_; g1v[6] = 0; g1v[7] = 0;

  unsigned long long gaq = (unsigned long long)(const char*)qb +
      ((unsigned long long)b * Tlen * K_ + (unsigned long long)h * HK_) * 4ull;
  unsigned long long gak = (unsigned long long)(const char*)kb +
      ((unsigned long long)b * Tlen * K_ + (unsigned long long)h * HK_) * 4ull;
  unsigned long long gag = (unsigned long long)(const char*)gkb +
      ((unsigned long long)b * Tlen * K_ + (unsigned long long)h * HK_) * 4ull;
  unsigned long long gav = (unsigned long long)(const char*)vb +
      ((unsigned long long)b * Tlen * V_ + (unsigned long long)h * HV_) * 4ull;
#endif

  for (int c0 = 0; c0 < Tlen; c0 += 16) {
    const size_t rbase = (size_t)b * (size_t)Tlen + (size_t)c0;

#if HAVE_TDM
    if (wave == 0) {
      tdm_load_2d((unsigned)(unsigned long long)&qe[0][0],  gaq, g1q);
      tdm_load_2d((unsigned)(unsigned long long)&keu[0][0], gak, g1k);
      tdm_load_2d((unsigned)(unsigned long long)&gcs[0][0], gag, g1k);
      tdm_load_2d((unsigned)(unsigned long long)&vch[0][0], gav, g1v);
      __builtin_amdgcn_s_wait_tensorcnt(0);
    }
#else
    for (int i = tid; i < 512; i += 256) {
      const int r = i >> 5, j = i & 31;
      const size_t off = (rbase + r) * (size_t)K_ + (size_t)h * HK_ + j;
      qe[r][j]  = qb[off];
      keu[r][j] = kb[off];
      gcs[r][j] = gkb[off];
    }
    for (int i = tid; i < 1024; i += 256) {
      const int r = i >> 6, j = i & 63;
      vch[r][j] = vb[(rbase + r) * (size_t)V_ + (size_t)h * HV_ + j];
    }
#endif
    __syncthreads();

    // ---- column-wise inclusive cumsum of gates ----
    if (tid < 32) {
      float run = 0.f;
#pragma unroll
      for (int r = 0; r < 16; ++r) { run += gcs[r][tid]; gcs[r][tid] = run; }
      eTot[tid] = __expf(run);
    }
    __syncthreads();

    // ---- build decorated q/k tensors ----
    for (int i = tid; i < 512; i += 256) {
      const int r = i >> 5, j = i & 31;
      const float gc = gcs[r][j];
      const float gt = gcs[15][j];
      const float qv = qe[r][j] * __expf(gc) * SCALE_;
      qe[r][j]    = qv;
      qe_bf[r][j] = (__bf16)qv;
      const float kraw = keu[r][j];
      ke_bf[r][j] = (__bf16)(kraw * __expf(-gc));
      keu[r][j]   = kraw * __expf(gt - gc);
    }
    __syncthreads();

    // ---- intra-chunk scores A = Qe @ Ke^T (one bf16 WMMA), causal mask ----
    if (wave == 0) {
      const int m   = lane & 15;
      const int kb0 = (lane < 16) ? 0 : 8;
      v16bf af  = concat8(*(const v8bf*)&qe_bf[m][kb0],
                          *(const v8bf*)&qe_bf[m][kb0 + 16]);
      v16bf bf2 = concat8(*(const v8bf*)&ke_bf[m][kb0],      // B col n = ke row
                          *(const v8bf*)&ke_bf[m][kb0 + 16]);
      v8f amat = {};
      amat = __builtin_amdgcn_wmma_f32_16x16x32_bf16(
          false, af, false, bf2, (short)0, amat, false, false);
      const int rb = (lane < 16) ? 0 : 8;
#pragma unroll
      for (int r2 = 0; r2 < 8; ++r2) {
        const int row = rb + r2;
        a_s[row][m] = (m <= row) ? amat[r2] : 0.f;   // causal, incl. diagonal
      }
    }
    __syncthreads();

    // ---- o = A @ V + Qe @ S_prev  (fp32 WMMA), waves 0..3 own 16-col slabs
    if (wave < 4) {
      v8f oacc = {};
      const int m   = lane & 15;
      const int kb2 = (lane < 16) ? 0 : 2;
      const int n   = wave * 16 + m;
#pragma unroll
      for (int kk = 0; kk < 4; ++kk) {        // A @ V  (K = 16)
        v2f a2 = *(const v2f*)&a_s[m][kk * 4 + kb2];
        v2f b2;
        b2.x = vch[kk * 4 + kb2][n];     b2.y = vch[kk * 4 + kb2 + 1][n];
        oacc = __builtin_amdgcn_wmma_f32_16x16x4_f32(
            false, a2, false, b2, (short)0, oacc, false, false);
      }
#pragma unroll
      for (int kk = 0; kk < 8; ++kk) {        // Qe @ S_prev  (K = 32)
        v2f a2 = *(const v2f*)&qe[m][kk * 4 + kb2];
        v2f b2;
        b2.x = S_s[kk * 4 + kb2][n];     b2.y = S_s[kk * 4 + kb2 + 1][n];
        oacc = __builtin_amdgcn_wmma_f32_16x16x4_f32(
            false, a2, false, b2, (short)0, oacc, false, false);
      }
      const int rb = (lane < 16) ? 0 : 8;
#pragma unroll
      for (int r2 = 0; r2 < 8; ++r2)
        ob[(rbase + rb + r2) * (size_t)V_ + (size_t)h * HV_ + n] = oacc[r2];
    }

    // ---- state update: S = S*exp(Gtot) + Keu^T @ V (all 8 waves) ----
    {
      const int rb = (lane < 16) ? 0 : 8;
#pragma unroll
      for (int r2 = 0; r2 < 8; ++r2) sacc[r2] *= eTot[mt * 16 + rb + r2];
      const int m2  = mt * 16 + (lane & 15);   // k-dim row of S
      const int n2  = nt * 16 + (lane & 15);   // v-dim col of S
      const int kb2 = (lane < 16) ? 0 : 2;
#pragma unroll
      for (int kk = 0; kk < 4; ++kk) {         // K = chunk time = 16
        v2f a2, b2;
        a2.x = keu[kk * 4 + kb2][m2];    a2.y = keu[kk * 4 + kb2 + 1][m2];
        b2.x = vch[kk * 4 + kb2][n2];    b2.y = vch[kk * 4 + kb2 + 1][n2];
        sacc = __builtin_amdgcn_wmma_f32_16x16x4_f32(
            false, a2, false, b2, (short)0, sacc, false, false);
      }
    }
    __syncthreads();                 // everyone done reading S_s / chunk LDS
    {
      const int rb = (lane < 16) ? 0 : 8;
      const int n2 = nt * 16 + (lane & 15);
#pragma unroll
      for (int r2 = 0; r2 < 8; ++r2) S_s[mt * 16 + rb + r2][n2] = sacc[r2];
    }
    __syncthreads();                 // S_s valid for next chunk

#if HAVE_TDM
    gaq += 16ull * K_ * 4ull;
    gak += 16ull * K_ * 4ull;
    gag += 16ull * K_ * 4ull;
    gav += 16ull * V_ * 4ull;
#endif
  }
}

// ---------------------------------------------------------------------------
// RMSNorm(o) * w * g*sigmoid(g). One block per (token, head), 64 threads.
// ---------------------------------------------------------------------------
__global__ __launch_bounds__(64) void gate_norm_kernel(
    const float* __restrict__ ob, const float* __restrict__ gb,
    const float* __restrict__ w, float* __restrict__ out)
{
  const int grp = blockIdx.x;          // token*H + h
  const int tid = threadIdx.x;
  const int token = grp / H_;
  const int h     = grp % H_;
  const size_t off = (size_t)token * V_ + (size_t)h * HV_ + tid;

  const float o = ob[off];
  float ss = o * o;
#pragma unroll
  for (int d = 16; d; d >>= 1) ss += __shfl_xor(ss, d, 32);
  __shared__ float part[2];
  if ((tid & 31) == 0) part[tid >> 5] = ss;
  __syncthreads();
  const float tot = part[0] + part[1];

  const float r  = rsqrtf(tot * (1.0f / HV_) + EPS_);
  const float g  = gb[off];
  const float sg = g / (1.0f + __expf(-g));      // g * sigmoid(g)
  out[off] = o * r * w[tid] * sg;
}

// ---------------------------------------------------------------------------
// host launcher
// ---------------------------------------------------------------------------
extern "C" void kernel_launch(void* const* d_in, const int* in_sizes, int n_in,
                              void* d_out, int out_size, void* d_ws, size_t ws_size,
                              hipStream_t stream) {
  (void)in_sizes; (void)n_in; (void)out_size; (void)ws_size;

  const float* x     = (const float*)d_in[0];
  const float* Wq    = (const float*)d_in[1];
  const float* Wk    = (const float*)d_in[2];
  const float* Wv    = (const float*)d_in[3];
  const float* Wgk1  = (const float*)d_in[4];
  const float* Wgk2  = (const float*)d_in[5];
  const float* bgk2  = (const float*)d_in[6];
  const float* Wg    = (const float*)d_in[7];
  const float* onw   = (const float*)d_in[8];
  const float* Wo    = (const float*)d_in[9];
  float* out = (float*)d_out;

  // workspace carve-up (fp32): q,k,gk [N,384]; v,g,o [N,768]; gated reuses v
  float* qbuf  = (float*)d_ws;
  float* kbuf  = qbuf  + (size_t)NTOK * K_;
  float* gkbuf = kbuf  + (size_t)NTOK * K_;
  float* vbuf  = gkbuf + (size_t)NTOK * K_;
  float* gbuf  = vbuf  + (size_t)NTOK * V_;
  float* obuf  = gbuf  + (size_t)NTOK * V_;
  float* gated = vbuf;   // v is dead after the scan

  const dim3 blk(256);

  // projections (bf16 WMMA GEMMs)
  gemm_bf16_wmma<<<dim3(K_ / 128, NTOK / 64), blk, 0, stream>>>(x, Wq, qbuf, NTOK, K_, D_, 1.0f);
  gemm_bf16_wmma<<<dim3(K_ / 128, NTOK / 64), blk, 0, stream>>>(x, Wk, kbuf, NTOK, K_, D_, 1.0f);
  gemm_bf16_wmma<<<dim3(V_ / 128, NTOK / 64), blk, 0, stream>>>(x, Wv, vbuf, NTOK, V_, D_, 1.0f);
  gemm_bf16_wmma<<<dim3(V_ / 128, NTOK / 64), blk, 0, stream>>>(x, Wg, gbuf, NTOK, V_, D_, 1.0f);

  // low-rank forget gates
  gk_lowrank_kernel<<<NTOK, blk, 0, stream>>>(x, Wgk1, Wgk2, bgk2, gkbuf);

  // chunked gated-linear-attention scan
  gla_scan_kernel<<<B_ * H_, blk, 0, stream>>>(qbuf, kbuf, vbuf, gkbuf, obuf, T_);

  // RMSNorm + swish output gate
  gate_norm_kernel<<<NTOK * H_, 64, 0, stream>>>(obuf, gbuf, onw, gated);

  // output projection
  gemm_bf16_wmma<<<dim3(D_ / 128, NTOK / 64), blk, 0, stream>>>(gated, Wo, out, NTOK, D_, V_, 1.0f);
}
